// QLSTMGen_65481071401780
// MI455X (gfx1250) — matrix operand order repaired and checked by reading, not compiled, not run on hardware
//
#include <hip/hip_runtime.h>
#include <hip/hip_bf16.h>

// ---------- types ----------
typedef __bf16 bf16_t;
typedef __attribute__((ext_vector_type(16))) __bf16 bf16x16;
typedef __attribute__((ext_vector_type(8)))  __bf16 bf16x8;
typedef __attribute__((ext_vector_type(8)))  float  f32x8;

#define TT 64
#define BB 256
#define DD 512
#define HH 512
#define NQ 8

__device__ __forceinline__ bf16_t f2bf(float f) {
  unsigned u = __float_as_uint(f);
  u += 0x7fffu + ((u >> 16) & 1u);          // round-to-nearest-even
  unsigned short h = (unsigned short)(u >> 16);
  bf16_t r;
  __builtin_memcpy(&r, &h, sizeof(r));
  return r;
}

__device__ __forceinline__ float sigmoidf(float x) {
  return 1.0f / (1.0f + __expf(-x));
}

// ---------- WMMA fragment loaders (CDNA5 16x16x32 bf16 layouts) ----------
// A (16x32, row-major [row][K]): lanes 0-15 -> K {0..7,16..23}; lanes 16-31 -> +8
__device__ __forceinline__ bf16x16 load_a_frag(const bf16_t* base, int row, int ld,
                                               int k0, int lane) {
  const int koff = (lane & 16) ? 8 : 0;
  const bf16_t* p = base + (size_t)row * ld + k0 + koff;
  bf16x8 lo = *(const bf16x8*)(p);
  bf16x8 hi = *(const bf16x8*)(p + 16);
  bf16x16 r;
#pragma unroll
  for (int i = 0; i < 8; ++i) { r[i] = lo[i]; r[i + 8] = hi[i]; }
  return r;
}
// B (32x16, stored column-major [col][K]): lanes 0-15 -> K {0..7,16..23}; lanes 16-31 -> +8
__device__ __forceinline__ bf16x16 load_b_frag(const bf16_t* base, int col, int ld,
                                               int k0, int lane) {
  const int koff = (lane & 16) ? 8 : 0;
  const bf16_t* p = base + (size_t)col * ld + k0 + koff;
  bf16x8 lo = *(const bf16x8*)(p);
  bf16x8 hi = *(const bf16x8*)(p + 16);
  bf16x16 r;
#pragma unroll
  for (int i = 0; i < 8; ++i) { r[i] = lo[i]; r[i + 8] = hi[i]; }
  return r;
}

// ---------- bf16 WMMA GEMM: D = A(MxK) * B^T(NxK) + biasN + Cadd ----------
// One wave computes a 16 x (16*NBLK) strip: A fragment reused across NBLK
// independent accumulators (no WMMA->WMMA RAW hazards, overlaps with loads).
// Grid must supply exactly (M/16)*(N/(16*NBLK)) waves.
template <int NBLK>
__global__ void k_gemm_bf16(const bf16_t* __restrict__ A, const bf16_t* __restrict__ B,
                            const float* __restrict__ biasN, const float* __restrict__ Cadd,
                            float* __restrict__ D, int M, int N, int K) {
  const int lane = threadIdx.x & 31;
  const int wave = (blockIdx.x * blockDim.x + threadIdx.x) >> 5;
  const int groupsN = N / (16 * NBLK);
  const int tm = wave / groupsN;
  const int tg = wave - tm * groupsN;
  if (tm >= (M >> 4)) return;                 // wave-uniform guard (EXEC stays all-1)
  const int m0 = tm << 4;
  const int n0 = tg * (16 * NBLK);
  const int rowA = m0 + (lane & 15);
  const int mBase = m0 + ((lane & 16) ? 8 : 0);

  f32x8 acc[NBLK];
#pragma unroll
  for (int j = 0; j < NBLK; ++j) {
    const int nIdx = n0 + j * 16 + (lane & 15);
    float binit = biasN ? biasN[nIdx] : 0.0f;
#pragma unroll
    for (int r = 0; r < 8; ++r) acc[j][r] = binit;
    if (Cadd) {
#pragma unroll
      for (int r = 0; r < 8; ++r) acc[j][r] += Cadd[(size_t)(mBase + r) * N + nIdx];
    }
  }

  for (int k0 = 0; k0 < K; k0 += 32) {
    bf16x16 a = load_a_frag(A, rowA, K, k0, lane);
    bf16x16 b[NBLK];
#pragma unroll
    for (int j = 0; j < NBLK; ++j)
      b[j] = load_b_frag(B, n0 + j * 16 + (lane & 15), K, k0, lane);
#pragma unroll
    for (int j = 0; j < NBLK; ++j)
      acc[j] = __builtin_amdgcn_wmma_f32_16x16x32_bf16(false, a, false, b[j],
                                                       (short)0, acc[j], false, false);
  }
#pragma unroll
  for (int j = 0; j < NBLK; ++j) {
    const int nIdx = n0 + j * 16 + (lane & 15);
#pragma unroll
    for (int r = 0; r < 8; ++r) D[(size_t)(mBase + r) * N + nIdx] = acc[j][r];
  }
}

// ---------- quantum statevector, one wave32 per (gate,batch) ----------
// amp index a = r*32 + lane : bits 0-4 = lane, bits 5-7 = register index r.
// wire q acts on bit p = 7-q (MSB-first reshape in the reference).
__device__ __forceinline__ void rx_gate(float (&re)[8], float (&im)[8], int p,
                                        float c, float s, int lane) {
  if (p >= 5) {
    const int m = 1 << (p - 5);
#pragma unroll
    for (int r = 0; r < 8; ++r)
      if (!(r & m)) {
        const int r1 = r | m;
        float re0 = re[r], im0 = im[r], re1 = re[r1], im1 = im[r1];
        re[r]  = c * re0 + s * im1;  im[r]  = c * im0 - s * re1;
        re[r1] = s * im0 + c * re1;  im[r1] = -s * re0 + c * im1;
      }
  } else {
    const int m = 1 << p;
    const bool hi = (lane & m) != 0;
#pragma unroll
    for (int r = 0; r < 8; ++r) {
      float pre = __shfl_xor(re[r], m, 32);
      float pim = __shfl_xor(im[r], m, 32);
      float nre, nim;
      if (!hi) { nre = c * re[r] + s * pim;  nim = c * im[r] - s * pre; }
      else     { nre = s * pim + c * re[r];  nim = -s * pre + c * im[r]; }
      re[r] = nre; im[r] = nim;
    }
  }
}

__device__ __forceinline__ void ry_gate(float (&re)[8], float (&im)[8], int p,
                                        float c, float s, int lane) {
  if (p >= 5) {
    const int m = 1 << (p - 5);
#pragma unroll
    for (int r = 0; r < 8; ++r)
      if (!(r & m)) {
        const int r1 = r | m;
        float re0 = re[r], im0 = im[r], re1 = re[r1], im1 = im[r1];
        re[r]  = c * re0 - s * re1;  im[r]  = c * im0 - s * im1;
        re[r1] = s * re0 + c * re1;  im[r1] = s * im0 + c * im1;
      }
  } else {
    const int m = 1 << p;
    const float ss = ((lane & m) != 0) ? s : -s;
#pragma unroll
    for (int r = 0; r < 8; ++r) {
      float pre = __shfl_xor(re[r], m, 32);
      float pim = __shfl_xor(im[r], m, 32);
      re[r] = c * re[r] + ss * pre;
      im[r] = c * im[r] + ss * pim;
    }
  }
}

__device__ __forceinline__ void rz_gate(float (&re)[8], float (&im)[8], int p,
                                        float c, float s, int lane) {
#pragma unroll
  for (int r = 0; r < 8; ++r) {
    const int bit = (p >= 5) ? ((r >> (p - 5)) & 1) : ((lane >> p) & 1);
    const float t = bit ? s : -s;
    float nre = c * re[r] - t * im[r];
    float nim = c * im[r] + t * re[r];
    re[r] = nre; im[r] = nim;
  }
}

__device__ __forceinline__ void cnot_gate(float (&re)[8], float (&im)[8], int pc,
                                          int pt, int lane) {
  if (pt >= 5) {                       // both bits in-register
    const int mc = 1 << (pc - 5), mt = 1 << (pt - 5);
#pragma unroll
    for (int r = 0; r < 8; ++r)
      if ((r & mc) && !(r & mt)) {
        const int r1 = r | mt;
        float tr = re[r]; re[r] = re[r1]; re[r1] = tr;
        float ti = im[r]; im[r] = im[r1]; im[r1] = ti;
      }
  } else {                             // target crosses lanes
    const int mt = 1 << pt;
#pragma unroll
    for (int r = 0; r < 8; ++r) {
      const int cb = (pc >= 5) ? ((r >> (pc - 5)) & 1) : ((lane >> pc) & 1);
      float pre = __shfl_xor(re[r], mt, 32);
      float pim = __shfl_xor(im[r], mt, 32);
      if (cb) { re[r] = pre; im[r] = pim; }
    }
  }
}

__global__ void k_quantum(const float* __restrict__ v,      // (B,32) angles
                          const float* __restrict__ qpf, const float* __restrict__ qpi,
                          const float* __restrict__ qpu, const float* __restrict__ qpo,
                          float* __restrict__ z) {           // (4,B,8)
  const int lane = threadIdx.x & 31;
  const int wave = (blockIdx.x * blockDim.x + threadIdx.x) >> 5;
  const int g = wave >> 8;            // gate 0..3
  const int b = wave & 255;
  const float* qp = (g == 0) ? qpf : (g == 1) ? qpi : (g == 2) ? qpu : qpo;

  float re[8], im[8];
#pragma unroll
  for (int r = 0; r < 8; ++r) { re[r] = 0.f; im[r] = 0.f; }
  re[0] = (lane == 0) ? 1.f : 0.f;

  const float* vrow = v + b * 32 + g * 8;
#pragma unroll
  for (int q = 0; q < NQ; ++q) {
    float th = vrow[q] * 0.5f;
    rx_gate(re, im, 7 - q, __cosf(th), __sinf(th), lane);
  }
  for (int l = 0; l < 2; ++l) {
#pragma unroll
    for (int q = 0; q < NQ; ++q) {
      float thy = qp[(l * NQ + q) * 3 + 0] * 0.5f;
      ry_gate(re, im, 7 - q, __cosf(thy), __sinf(thy), lane);
      float thz = qp[(l * NQ + q) * 3 + 1] * 0.5f;
      rz_gate(re, im, 7 - q, __cosf(thz), __sinf(thz), lane);
    }
#pragma unroll
    for (int q = 0; q < NQ - 1; ++q) cnot_gate(re, im, 7 - q, 6 - q, lane);
  }

  float zq[8];
#pragma unroll
  for (int q = 0; q < NQ; ++q) zq[q] = 0.f;
#pragma unroll
  for (int r = 0; r < 8; ++r) {
    float prob = re[r] * re[r] + im[r] * im[r];
    const int a = (r << 5) | lane;
#pragma unroll
    for (int q = 0; q < NQ; ++q)
      zq[q] += ((a >> (7 - q)) & 1) ? -prob : prob;
  }
#pragma unroll
  for (int q = 0; q < NQ; ++q) {
#pragma unroll
    for (int off = 16; off >= 1; off >>= 1) zq[q] += __shfl_xor(zq[q], off, 32);
  }
  if (lane == 0) {
#pragma unroll
    for (int q = 0; q < NQ; ++q) z[(g * BB + b) * NQ + q] = zq[q];
  }
}

// ---------- h1 = relu(z @ pW1 + pb1) -> bf16  (K=8, scalar FMAs) ----------
__global__ void k_h1(const float* __restrict__ z, const float* __restrict__ pW1,
                     const float* __restrict__ pb1, bf16_t* __restrict__ h1) {
  const int idx = blockIdx.x * blockDim.x + threadIdx.x;   // 4*B*H
  const int h = idx & (HH - 1);
  const int row = idx >> 9;
  const float* zr = z + row * NQ;
  float acc = pb1[h];
#pragma unroll
  for (int q = 0; q < NQ; ++q) acc += zr[q] * pW1[q * HH + h];
  h1[idx] = f2bf(fmaxf(acc, 0.f));
}

// ---------- LSTM elementwise update ----------
__global__ void k_lstm(const float* __restrict__ y, float* __restrict__ cx,
                       bf16_t* __restrict__ hx_bf, float* __restrict__ out_t,
                       float* __restrict__ out_h, float* __restrict__ out_c, int last) {
  const int idx = blockIdx.x * blockDim.x + threadIdx.x;   // B*H
  float f = sigmoidf(y[(0 * BB * HH) + idx]);
  float i = sigmoidf(y[(1 * BB * HH) + idx]);
  float g = tanhf   (y[(2 * BB * HH) + idx]);
  float o = sigmoidf(y[(3 * BB * HH) + idx]);
  float c = f * cx[idx] + i * g;
  float hn = o * tanhf(c);
  cx[idx] = c;
  hx_bf[idx] = f2bf(hn);
  out_t[idx] = hn;
  if (last) { out_h[idx] = hn; out_c[idx] = c; }
}

// ---------- one-time prep: weight repack/convert, state zeroing ----------
__global__ void k_prep(const float* __restrict__ Wf, const float* __restrict__ Wi,
                       const float* __restrict__ Wu, const float* __restrict__ Wo,
                       const float* __restrict__ bfv, const float* __restrict__ biv,
                       const float* __restrict__ buv, const float* __restrict__ bov,
                       const float* __restrict__ pW2,
                       bf16_t* __restrict__ WxT, bf16_t* __restrict__ WhT,
                       bf16_t* __restrict__ pW2T, float* __restrict__ bias_cat,
                       float* __restrict__ cx, bf16_t* __restrict__ hx_bf) {
  const int idx = blockIdx.x * blockDim.x + threadIdx.x;   // 512*512 threads
  if (idx < HH * HH) {                                     // pW2T[n][k] = pW2[k][n]
    const int n = idx >> 9, k = idx & (HH - 1);
    pW2T[idx] = f2bf(pW2[k * HH + n]);
  }
  if (idx < 32 * DD) {                                     // W*T[g*8+n][k]
    const int row = idx >> 9, k = idx & (DD - 1);
    const int g = row >> 3, n = row & 7;
    const float* W = (g == 0) ? Wf : (g == 1) ? Wi : (g == 2) ? Wu : Wo;
    WxT[idx] = f2bf(W[k * NQ + n]);
    WhT[idx] = f2bf(W[(DD + k) * NQ + n]);
  }
  if (idx < 32) {
    const int g = idx >> 3, n = idx & 7;
    const float* bb = (g == 0) ? bfv : (g == 1) ? biv : (g == 2) ? buv : bov;
    bias_cat[idx] = bb[n];
  }
  if (idx < BB * HH) { cx[idx] = 0.f; hx_bf[idx] = f2bf(0.f); }
}

__global__ void k_cvt_x(const float* __restrict__ x, bf16_t* __restrict__ xb, int n) {
  const int idx = blockIdx.x * blockDim.x + threadIdx.x;
  if (idx < n) xb[idx] = f2bf(x[idx]);
}

// ---------- host launch ----------
extern "C" void kernel_launch(void* const* d_in, const int* in_sizes, int n_in,
                              void* d_out, int out_size, void* d_ws, size_t ws_size,
                              hipStream_t stream) {
  (void)in_sizes; (void)n_in; (void)out_size; (void)ws_size;
  const float* x    = (const float*)d_in[0];
  const float* qpf  = (const float*)d_in[1];
  const float* qpi  = (const float*)d_in[2];
  const float* qpu  = (const float*)d_in[3];
  const float* qpo  = (const float*)d_in[4];
  const float* Wf   = (const float*)d_in[5];
  const float* bfv  = (const float*)d_in[6];
  const float* Wi   = (const float*)d_in[7];
  const float* biv  = (const float*)d_in[8];
  const float* Wu   = (const float*)d_in[9];
  const float* buv  = (const float*)d_in[10];
  const float* Wo   = (const float*)d_in[11];
  const float* bov  = (const float*)d_in[12];
  const float* pW1  = (const float*)d_in[13];
  const float* pb1  = (const float*)d_in[14];
  const float* pW2  = (const float*)d_in[15];
  const float* pb2  = (const float*)d_in[16];
  float* out = (float*)d_out;

  // workspace carve-up (256B aligned)
  char* w = (char*)d_ws;
  size_t off = 0;
  auto carve = [&](size_t bytes) {
    void* p = w + off;
    off = (off + bytes + 255) & ~(size_t)255;
    return p;
  };
  bf16_t* x_bf     = (bf16_t*)carve((size_t)TT * BB * DD * 2);
  float*  xproj    = (float*) carve((size_t)TT * BB * 32 * 4);
  bf16_t* WxT      = (bf16_t*)carve(32 * DD * 2);
  bf16_t* WhT      = (bf16_t*)carve(32 * HH * 2);
  bf16_t* pW2T     = (bf16_t*)carve(HH * HH * 2);
  float*  bias_cat = (float*) carve(32 * 4);
  float*  v        = (float*) carve(BB * 32 * 4);
  float*  z        = (float*) carve(4 * BB * NQ * 4);
  bf16_t* h1       = (bf16_t*)carve((size_t)4 * BB * HH * 2);
  float*  y        = (float*) carve((size_t)4 * BB * HH * 4);
  float*  cx       = (float*) carve((size_t)BB * HH * 4);
  bf16_t* hx_bf    = (bf16_t*)carve((size_t)BB * HH * 2);

  // one-time: repack weights, zero state, convert inputs, hoisted x-projection
  k_prep<<<1024, 256, 0, stream>>>(Wf, Wi, Wu, Wo, bfv, biv, buv, bov, pW2,
                                   WxT, WhT, pW2T, bias_cat, cx, hx_bf);
  const int nx = TT * BB * DD;
  k_cvt_x<<<(nx + 255) / 256, 256, 0, stream>>>(x, x_bf, nx);
  // xproj = x_bf (16384x512) @ WxT^T -> (16384x32):
  //   NBLK=2 -> 1024x1 wave-strips = 1024 waves = 128 blocks
  k_gemm_bf16<2><<<128, 256, 0, stream>>>(x_bf, WxT, nullptr, nullptr, xproj,
                                          TT * BB, 32, DD);

  for (int t = 0; t < TT; ++t) {
    // v = hx @ WhT^T + xproj[t] + bias : M=256,N=32,NBLK=2 -> 16 waves = 2 blocks
    k_gemm_bf16<2><<<2, 256, 0, stream>>>(hx_bf, WhT, bias_cat,
                                          xproj + (size_t)t * BB * 32, v, BB, 32, HH);
    // 1024 wave32 statevector sims
    k_quantum<<<128, 256, 0, stream>>>(v, qpf, qpi, qpu, qpo, z);
    // h1 = relu(z @ pW1 + pb1)
    k_h1<<<(4 * BB * HH) / 256, 256, 0, stream>>>(z, pW1, pb1, h1);
    // y = h1 (1024x512) @ pW2T^T + pb2 : M=1024,N=512,NBLK=4 -> 512 waves = 64 blocks
    k_gemm_bf16<4><<<64, 256, 0, stream>>>(h1, pW2T, pb2, nullptr, y, 4 * BB, HH, HH);
    // gating + state update + outputs[t]
    k_lstm<<<(BB * HH) / 256, 256, 0, stream>>>(
        y, cx, hx_bf, out + (size_t)t * BB * HH,
        out + (size_t)TT * BB * HH, out + (size_t)TT * BB * HH + (size_t)BB * HH,
        (t == TT - 1) ? 1 : 0);
  }
}
